// resCNN_34926674051581
// MI455X (gfx1250) — matrix operand order repaired and checked by reading, not compiled, not run on hardware
//
#include <hip/hip_runtime.h>
#include <hip/hip_bf16.h>
#include <math.h>

// ---------------------------------------------------------------------------
// MI455X (gfx1250) locally-connected CNN, f16 WMMA pipeline.
//   x[4096,6,60,60] -> LC(6->32,3x3) tanh -> LC(32->64,2x2) tanh
//                   -> LC(64->128,5x5) tanh -> concat info -> Linear(514,2) -> softmax
// Each LC layer = independent per-location GEMMs mapped to v_wmma_f32_16x16x32_f16.
// Epilogue tanh: branch-free, native v_exp_f32 + v_rcp_f32 (no IEEE div expansion).
// ---------------------------------------------------------------------------

typedef __attribute__((ext_vector_type(16))) _Float16 v16h;
typedef __attribute__((ext_vector_type(8)))  _Float16 v8h;
typedef __attribute__((ext_vector_type(8)))  float    v8f;

#define BATCH 4096

__device__ __forceinline__ v16h frag2(const _Float16* p0, const _Float16* p1) {
  union { v16h v; v8h h[2]; } u;
  u.h[0] = *(const v8h*)p0;
  u.h[1] = *(const v8h*)p1;
  return u.v;
}

__device__ __forceinline__ v8f wmma_f16(v16h a, v16h b, v8f c) {
  // 8 args: (neg_a, A, neg_b, B, c_mod, C, reuse_a, reuse_b)
  return __builtin_amdgcn_wmma_f32_16x16x32_f16(false, a, false, b, (short)0, c,
                                                false, false);
}

// Branch-free tanh: sign(x) * (1 - 2*rcp(exp(2|x|)+1)).
// exp overflow -> inf -> rcp(inf) = 0 -> saturates to +/-1. No EXEC divergence,
// no IEEE div expansion (v_exp_f32 + v_rcp_f32 only).
__device__ __forceinline__ float fast_tanh(float x) {
  float ax = __builtin_fabsf(x);
  float e  = __expf(ax + ax);                     // v_exp_f32
  float r  = __builtin_fmaf(-2.f, __builtin_amdgcn_rcpf(e + 1.f), 1.f);
  return __builtin_copysignf(r, x);
}

// ---------------------------------------------------------------------------
// Weight repacks: f32 -> f16, into B-fragment-friendly [loc][N][K] layout.
// ---------------------------------------------------------------------------

// W0 [20,20,32,6,3,3] -> W0p [400][32][64]  (K order (c,k,l) == source, pad 54->64)
__global__ void repack_w0(const float* __restrict__ W0, _Float16* __restrict__ Wp) {
  int tid = blockIdx.x * blockDim.x + threadIdx.x;       // 400*32*64 = 819200
  int loc = tid >> 11;            // /(32*64)
  int r   = tid & 2047;
  int nq  = r >> 6;
  int kk  = r & 63;
  _Float16 v = (_Float16)0.f;
  if (kk < 54) v = (_Float16)W0[(size_t)(loc * 32 + nq) * 54 + kk];
  Wp[tid] = v;
}

// W1 [10,10,64,32,2,2] -> W1p [100][64][128]  with K reordered to (k,l,c)
__global__ void repack_w1(const float* __restrict__ W1, _Float16* __restrict__ Wp) {
  int tid = blockIdx.x * blockDim.x + threadIdx.x;       // 100*64*128 = 819200
  int loc = tid >> 13;            // /(64*128)
  int r   = tid & 8191;
  int nq  = r >> 7;
  int kk  = r & 127;
  int kl  = kk >> 5;              // (k*2+l)
  int c   = kk & 31;
  size_t src = (size_t)((loc * 64 + nq) * 32 + c) * 4 + kl;
  Wp[tid] = (_Float16)W1[src];
}

// W2 [2,2,128,64,5,5] -> W2p [4][128][1600]  with K reordered to (k,l,c)
__global__ void repack_w2(const float* __restrict__ W2, _Float16* __restrict__ Wp) {
  int tid = blockIdx.x * blockDim.x + threadIdx.x;       // 4*128*1600 = 819200
  int loc = tid / (128 * 1600);
  int r   = tid - loc * (128 * 1600);
  int nq  = r / 1600;
  int kk  = r - nq * 1600;
  int chunk = kk >> 6;            // (k*5+l)
  int c     = kk & 63;
  size_t src = (size_t)((loc * 128 + nq) * 64 + c) * 25 + chunk;
  Wp[tid] = (_Float16)W2[src];
}

// ---------------------------------------------------------------------------
// Layer 0: per-location GEMM M=4096,K=54(pad 64),N=32. LDS im2col staging.
// Block: (loc, mblock of 64 rows), 256 thr = 8 waves -> 4 mtiles x 2 ntiles.
// Output H0 [20*20][B][32] f16 (location-major).
// ---------------------------------------------------------------------------
__global__ void lc0_gemm(const float* __restrict__ x, const _Float16* __restrict__ Wp,
                         const float* __restrict__ bias, _Float16* __restrict__ H0) {
  const int loc = blockIdx.x;       // 0..399
  const int mblock = blockIdx.y;    // 0..63
  const int t = threadIdx.x;
  const int lane = t & 31, wave = t >> 5;
  const int i = loc / 20, j = loc % 20;

  __shared__ __align__(16) _Float16 sA[64 * 64];   // 8 KB: 64 rows x 64 K

  // cooperative im2col gather + f32->f16 convert (K order (c,k,l), pad zeros)
  #pragma unroll
  for (int e = 0; e < 16; ++e) {
    int idx = e * 256 + t;
    int r = idx >> 6, kk = idx & 63;
    _Float16 v = (_Float16)0.f;
    if (kk < 54) {
      int c = kk / 9, rem = kk - c * 9, kr = rem / 3, kl = rem - kr * 3;
      size_t b = (size_t)(mblock * 64 + r);
      v = (_Float16)x[((b * 6 + c) * 60 + (3 * i + kr)) * 60 + (3 * j + kl)];
    }
    sA[idx] = v;
  }
  __syncthreads();

  const int mtile = wave >> 1, ntile = wave & 1;
  const int row = lane & 15, grp = lane >> 4;
  v8f acc = {};
  const _Float16* Bp = Wp + ((size_t)loc * 32 + ntile * 16 + row) * 64;
  #pragma unroll
  for (int s = 0; s < 2; ++s) {
    const _Float16* ap = &sA[(mtile * 16 + row) * 64 + s * 32 + grp * 8];
    v16h a = frag2(ap, ap + 16);
    const _Float16* bp = Bp + s * 32 + grp * 8;
    v16h b = frag2(bp, bp + 16);
    acc = wmma_f16(a, b, acc);
  }
  const int ng = ntile * 16 + row;
  const float bs = bias[loc * 32 + ng];
  const int mBase = mblock * 64 + mtile * 16 + grp * 8;
  _Float16* outp = H0 + (size_t)loc * BATCH * 32;
  #pragma unroll
  for (int e = 0; e < 8; ++e) {
    float v = fast_tanh(acc[e] + bs);
    outp[(size_t)(mBase + e) * 32 + ng] = (_Float16)v;
  }
}

// ---------------------------------------------------------------------------
// Layer 1: per-location GEMM M=4096,K=128,N=64. A read directly from H0
// (each 32-wide K step = one spatial chunk, contiguous channels).
// Block: (loc, mblock of 32 rows), 8 waves -> 2 mtiles x 4 ntiles.
// Output H1 [10*10][B][64] f16.
// ---------------------------------------------------------------------------
__global__ void lc1_gemm(const _Float16* __restrict__ H0, const _Float16* __restrict__ Wp,
                         const float* __restrict__ bias, _Float16* __restrict__ H1) {
  const int loc = blockIdx.x;     // 0..99
  const int mblock = blockIdx.y;  // 0..127
  const int t = threadIdx.x, lane = t & 31, wave = t >> 5;
  const int mtile = wave >> 2, ntile = wave & 3;
  const int row = lane & 15, grp = lane >> 4;
  const int i = loc / 10, j = loc - i * 10;
  const int m0 = mblock * 32 + mtile * 16;

  v8f acc = {};
  const _Float16* Bp = Wp + ((size_t)loc * 64 + ntile * 16 + row) * 128;
  #pragma unroll
  for (int s = 0; s < 4; ++s) {
    const int kr = s >> 1, kl = s & 1;
    const int c0 = (2 * i + kr) * 20 + (2 * j + kl);
    const _Float16* ap = H0 + ((size_t)c0 * BATCH + m0 + row) * 32 + grp * 8;
    v16h a = frag2(ap, ap + 16);
    const _Float16* bp = Bp + s * 32 + grp * 8;
    v16h b = frag2(bp, bp + 16);
    acc = wmma_f16(a, b, acc);
  }
  const int ng = ntile * 16 + row;
  const float bs = bias[loc * 64 + ng];
  _Float16* outp = H1 + (size_t)loc * BATCH * 64;
  const int mBase = m0 + grp * 8;
  #pragma unroll
  for (int e = 0; e < 8; ++e) {
    float v = fast_tanh(acc[e] + bs);
    outp[(size_t)(mBase + e) * 64 + ng] = (_Float16)v;
  }
}

// ---------------------------------------------------------------------------
// Layer 2: per-location GEMM M=4096,K=1600,N=128.
// Block: (loc, mblock of 16 rows), 8 waves = 8 ntiles, 50 K-steps each.
// Output H2 [2*2][B][128] f16.
// ---------------------------------------------------------------------------
__global__ void lc2_gemm(const _Float16* __restrict__ H1, const _Float16* __restrict__ Wp,
                         const float* __restrict__ bias, _Float16* __restrict__ H2) {
  const int loc = blockIdx.x;       // 0..3
  const int m0 = blockIdx.y * 16;   // 0..255 -> batch row base
  const int t = threadIdx.x, lane = t & 31, ntile = t >> 5;
  const int row = lane & 15, grp = lane >> 4;
  const int i = loc >> 1, j = loc & 1;

  v8f acc = {};
  const _Float16* Bp = Wp + ((size_t)loc * 128 + ntile * 16 + row) * 1600;
  const _Float16* Arow = H1 + (size_t)(m0 + row) * 64 + grp * 8;
  #pragma unroll 2
  for (int s = 0; s < 50; ++s) {
    const int p = s >> 1, half = s & 1;
    const int kr = p / 5, kl = p - kr * 5;
    const int c0 = (5 * i + kr) * 10 + (5 * j + kl);
    const _Float16* ap = Arow + (size_t)c0 * BATCH * 64 + half * 32;
    v16h a = frag2(ap, ap + 16);
    const _Float16* bp = Bp + s * 32 + grp * 8;
    v16h b = frag2(bp, bp + 16);
    acc = wmma_f16(a, b, acc);
  }
  const int ng = ntile * 16 + row;
  const float bs = bias[loc * 128 + ng];
  _Float16* outp = H2 + (size_t)loc * BATCH * 128;
  const int mBase = m0 + grp * 8;
  #pragma unroll
  for (int e = 0; e < 8; ++e) {
    float v = fast_tanh(acc[e] + bs);
    outp[(size_t)(mBase + e) * 128 + ng] = (_Float16)v;
  }
}

// ---------------------------------------------------------------------------
// Classifier: [B,512] (+info[B,2]) @ Wc[514,2] + bc, then 2-way softmax.
// Flatten order of reference: feature f = o*4 + i*2 + j; H2 is [i*2+j][b][o].
// ---------------------------------------------------------------------------
__global__ void classify(const _Float16* __restrict__ H2, const float* __restrict__ info,
                         const float* __restrict__ Wc, const float* __restrict__ bc,
                         float* __restrict__ out) {
  const int b = blockIdx.x * blockDim.x + threadIdx.x;   // 0..4095
  float a0 = bc[0], a1 = bc[1];
  #pragma unroll
  for (int locc = 0; locc < 4; ++locc) {
    const int i = locc >> 1, j = locc & 1;
    const _Float16* hp = H2 + ((size_t)locc * BATCH + b) * 128;
    for (int o = 0; o < 128; ++o) {
      float v = (float)hp[o];
      int f = (o * 2 + i) * 2 + j;
      a0 += v * Wc[f * 2 + 0];
      a1 += v * Wc[f * 2 + 1];
    }
  }
  float i0 = info[b * 2 + 0], i1 = info[b * 2 + 1];
  a0 += i0 * Wc[512 * 2 + 0] + i1 * Wc[513 * 2 + 0];
  a1 += i0 * Wc[512 * 2 + 1] + i1 * Wc[513 * 2 + 1];
  float m = fmaxf(a0, a1);
  float e0 = __expf(a0 - m), e1 = __expf(a1 - m);
  float inv = __builtin_amdgcn_rcpf(e0 + e1);   // single v_rcp_f32
  out[b * 2 + 0] = e0 * inv;
  out[b * 2 + 1] = e1 * inv;
}

// ---------------------------------------------------------------------------
extern "C" void kernel_launch(void* const* d_in, const int* in_sizes, int n_in,
                              void* d_out, int out_size, void* d_ws, size_t ws_size,
                              hipStream_t stream) {
  (void)in_sizes; (void)n_in; (void)out_size; (void)ws_size;
  const float* x    = (const float*)d_in[0];
  const float* info = (const float*)d_in[1];
  const float* W0   = (const float*)d_in[2];
  const float* B0   = (const float*)d_in[3];
  const float* W1   = (const float*)d_in[4];
  const float* B1   = (const float*)d_in[5];
  const float* W2   = (const float*)d_in[6];
  const float* B2   = (const float*)d_in[7];
  const float* Wc   = (const float*)d_in[8];
  const float* bc   = (const float*)d_in[9];
  float* out = (float*)d_out;

  // Workspace carve-up (f16 buffers; total ~167 MB)
  char* ws = (char*)d_ws;
  _Float16* W0p = (_Float16*)ws; ws += (size_t)400 * 32 * 64 * 2;       // 1.6 MB
  _Float16* W1p = (_Float16*)ws; ws += (size_t)100 * 64 * 128 * 2;      // 1.6 MB
  _Float16* W2p = (_Float16*)ws; ws += (size_t)4 * 128 * 1600 * 2;      // 1.6 MB
  _Float16* H0  = (_Float16*)ws; ws += (size_t)400 * BATCH * 32 * 2;    // 104.9 MB
  _Float16* H1  = (_Float16*)ws; ws += (size_t)100 * BATCH * 64 * 2;    // 52.4 MB
  _Float16* H2  = (_Float16*)ws;                                        // 4.2 MB

  repack_w0<<<3200, 256, 0, stream>>>(W0, W0p);
  repack_w1<<<3200, 256, 0, stream>>>(W1, W1p);
  repack_w2<<<3200, 256, 0, stream>>>(W2, W2p);

  lc0_gemm<<<dim3(400, 64),  256, 0, stream>>>(x,  W0p, B0, H0);
  lc1_gemm<<<dim3(100, 128), 256, 0, stream>>>(H0, W1p, B1, H1);
  lc2_gemm<<<dim3(4, 256),   256, 0, stream>>>(H1, W2p, B2, H2);

  classify<<<16, 256, 0, stream>>>(H2, info, Wc, bc, out);
}
